// Insert_net_2_48223892800166
// MI455X (gfx1250) — matrix operand order repaired and checked by reading, not compiled, not run on hardware
//
#include <hip/hip_runtime.h>
#include <math.h>
#include <stdint.h>

// ---------------------------------------------------------------------------
// Types for CDNA5 WMMA
// ---------------------------------------------------------------------------
typedef __attribute__((ext_vector_type(16))) _Float16 v16h;
typedef __attribute__((ext_vector_type(8)))  float    v8f;

#define THREADS 256

// ---------------------------------------------------------------------------
// Problem constants (from reference)
// ---------------------------------------------------------------------------
#define BB 2
#define HH 256
#define WW_ 448
#define H2 (HH/2)
#define W2 (WW_/2)
#define H4 (HH/4)
#define W4 (WW_/4)

// ---------------------------------------------------------------------------
// Kernel: pack OIHW f32 weights into the CDNA5 16-bit A-fragment layout.
// Layout: [mtile][kc][lane(32)][half(16)] f16, K = (ky*3+kx)*Cinpad + ci.
// A-matrix 16x32 f16 lane layout (contiguous K, matches B):
//   lane<16 : M=lane,    half h -> K=h
//   lane>=16: M=lane-16, half h -> K=16+h
// ---------------------------------------------------------------------------
__global__ void pack_weights_k(const float* __restrict__ w, _Float16* __restrict__ dst,
                               int Cout, int Cin, int Cinpad, int KC, size_t total)
{
    size_t idx = (size_t)blockIdx.x * THREADS + threadIdx.x;
    if (idx >= total) return;
    int h    = (int)(idx & 15);
    int lane = (int)((idx >> 4) & 31);
    size_t t = idx >> 9;
    int kc = (int)(t % (size_t)KC);
    int mt = (int)(t / (size_t)KC);
    int v = h >> 1, j = h & 1;
    int m, kl;
    if (lane < 16) { m = lane;      kl = 2*v + j; }
    else           { m = lane - 16; kl = 16 + 2*v + j; }
    int kcpp = Cinpad >> 5;            // 32-wide K chunks per (ky,kx)
    int kpos = kc / kcpp;
    int cb   = kc - kpos * kcpp;
    int ci   = cb * 32 + kl;
    int ky   = kpos / 3;
    int kx   = kpos - 3 * ky;
    int co   = mt * 16 + m;
    float val = 0.f;
    if (co < Cout && ci < Cin)
        val = w[(((size_t)co * Cin + ci) * 3 + ky) * 3 + kx];
    dst[idx] = (_Float16)val;
}

// ---------------------------------------------------------------------------
// Kernel: NCHW f32 (channel slice) -> NHWC f16 (with channel padding)
// ---------------------------------------------------------------------------
__global__ void nchw_to_nhwc_f16_k(const float* __restrict__ src, int srcCtot, int srcCoff,
                                   int Ccopy, _Float16* __restrict__ dst, int Cpad, int dstCoff,
                                   int Bn, int HW)
{
    size_t idx = (size_t)blockIdx.x * THREADS + threadIdx.x;
    size_t total = (size_t)Bn * HW * Ccopy;
    if (idx >= total) return;
    int c = (int)(idx % (size_t)Ccopy);
    size_t p = idx / (size_t)Ccopy;
    int b  = (int)(p / (size_t)HW);
    int hw = (int)(p % (size_t)HW);
    dst[((size_t)b * HW + hw) * Cpad + dstCoff + c] =
        (_Float16)src[((size_t)b * srcCtot + srcCoff + c) * HW + hw];
}

// ---------------------------------------------------------------------------
// Kernel: 3x3 SAME conv, implicit GEMM via v_wmma_f32_16x16x32_f16.
// One wave computes a 16-cout x 16-pixel output tile.
//   A = packed weights (16 x K), B = NHWC activations (K x 16 pixels).
// B-fragment f16 lane layout: n = lane%16 (pixel), K = kbase + h,
//   kbase = (lane<16) ? 0 : 16 within the 32-wide K chunk
//   -> each lane loads 16 consecutive f16 channels (32 bytes) per chunk.
// lax SAME: pad_lo = 1 for stride 1, pad_lo = 0 for stride 2 (k=3).
// ---------------------------------------------------------------------------
__global__ __launch_bounds__(256) void conv3x3_wmma_k(
    const _Float16* __restrict__ in,   // [B, Hin, Win, Cinpad]
    const _Float16* __restrict__ wp,   // packed weights
    const float* __restrict__ bias,    // [Cout]
    float* __restrict__ out,           // [B, Cout, Hout, Wout]
    int Hin, int Win, int Hout, int Wout,
    int Cinpad, int Cout, int stride, int relu)
{
    const int lane = threadIdx.x & 31;
    const int wave = threadIdx.x >> 5;
    const int tilesX = Wout >> 4;
    const int tilesPerImg = (Hout * Wout) >> 4;
    const int tile = blockIdx.x * 8 + wave;
    if (tile >= tilesPerImg) return;          // wave-uniform
    const int mt = blockIdx.y;
    const int b  = blockIdx.z;
    const int ty = tile / tilesX;
    const int x  = ((tile - ty * tilesX) << 4) + (lane & 15);
    const int padlo = (stride == 1) ? 1 : 0;
    const int kcpp  = Cinpad >> 5;
    const int KC    = 9 * kcpp;
    const _Float16* a_ptr = wp + ((size_t)mt * KC * 32 + lane) * 16;
    const int ciBase = (lane & 16) ? 16 : 0;

    v8f acc = {};
    for (int kpos = 0; kpos < 9; ++kpos) {
        const int ky = kpos / 3;
        const int kx = kpos - 3 * ky;
        const int iy = ty * stride + ky - padlo;
        const int ix = x  * stride + kx - padlo;
        const bool ok = (iy >= 0) && (iy < Hin) && (ix >= 0) && (ix < Win);
        const _Float16* bp = in + (((size_t)b * Hin + (ok ? iy : 0)) * Win + (ok ? ix : 0)) * Cinpad + ciBase;
        for (int cb = 0; cb < kcpp; ++cb) {
            v16h af = *(const v16h*)a_ptr;
            a_ptr += 512;                      // next K-chunk: 32 lanes * 16 halves
            v16h bf = {};
            if (ok) bf = *(const v16h*)(bp + (size_t)cb * 32);
            acc = __builtin_amdgcn_wmma_f32_16x16x32_f16(
                      false, af, false, bf, (short)0, acc, false, false);
        }
    }

    // C/D layout: VGPR r holds M = (lane<16 ? r : 8+r), N = lane%16
    const int mrow = (lane & 16) ? 8 : 0;
#pragma unroll
    for (int r = 0; r < 8; ++r) {
        int co = mt * 16 + mrow + r;
        if (co < Cout) {
            float v = acc[r] + bias[co];
            if (relu) v = fmaxf(v, 0.f);
            out[(((size_t)b * Cout + co) * Hout + ty) * Wout + x] = v;
        }
    }
}

// ---------------------------------------------------------------------------
// Kernel: bilinear resize (half-pixel centers, edge clamp), times a scalar.
// ---------------------------------------------------------------------------
__global__ void resize_bilinear_k(const float* __restrict__ src, float* __restrict__ dst,
                                  int C, int Hs, int Ws, int Hd, int Wd, int Bn, float mul)
{
    size_t idx = (size_t)blockIdx.x * THREADS + threadIdx.x;
    size_t total = (size_t)Bn * C * Hd * Wd;
    if (idx >= total) return;
    int xd = (int)(idx % (size_t)Wd);
    size_t t = idx / (size_t)Wd;
    int yd = (int)(t % (size_t)Hd); t /= (size_t)Hd;
    int c  = (int)(t % (size_t)C);
    int b  = (int)(t / (size_t)C);
    float sx = (xd + 0.5f) * ((float)Ws / (float)Wd) - 0.5f;
    float sy = (yd + 0.5f) * ((float)Hs / (float)Hd) - 0.5f;
    float fxf = floorf(sx), fyf = floorf(sy);
    int x0 = (int)fxf, y0 = (int)fyf;
    float gx = sx - fxf, gy = sy - fyf;
    int x1 = x0 + 1, y1 = y0 + 1;
    x0 = min(max(x0, 0), Ws - 1); x1 = min(max(x1, 0), Ws - 1);
    y0 = min(max(y0, 0), Hs - 1); y1 = min(max(y1, 0), Hs - 1);
    const float* s = src + ((size_t)b * C + c) * Hs * Ws;
    float v00 = s[(size_t)y0 * Ws + x0], v01 = s[(size_t)y0 * Ws + x1];
    float v10 = s[(size_t)y1 * Ws + x0], v11 = s[(size_t)y1 * Ws + x1];
    float v = v00 * (1.f - gx) * (1.f - gy) + v01 * gx * (1.f - gy)
            + v10 * (1.f - gx) * gy         + v11 * gx * gy;
    dst[idx] = mul * v;
}

// ---------------------------------------------------------------------------
// Kernel: backwarp img by flow, output channel-mean |ref - warped| (3 ch)
// ---------------------------------------------------------------------------
__global__ void warp_absdiff_k(const float* __restrict__ ref, const float* __restrict__ img,
                               const float* __restrict__ flow, float* __restrict__ outM,
                               int Hh, int Ww, int Bn)
{
    size_t idx = (size_t)blockIdx.x * THREADS + threadIdx.x;
    int HW = Hh * Ww;
    size_t total = (size_t)Bn * HW;
    if (idx >= total) return;
    int b  = (int)(idx / (size_t)HW);
    int hw = (int)(idx % (size_t)HW);
    int y = hw / Ww, x = hw - y * Ww;
    float fx = (float)x + flow[((size_t)b * 2 + 0) * HW + hw];
    float fy = (float)y + flow[((size_t)b * 2 + 1) * HW + hw];
    float flx = floorf(fx), fly = floorf(fy);
    int x0 = (int)flx, y0 = (int)fly;
    float gx = fx - flx, gy = fy - fly;
    int   xs[4] = { x0, x0 + 1, x0,     x0 + 1 };
    int   ys[4] = { y0, y0,     y0 + 1, y0 + 1 };
    float tw[4] = { (1.f-gx)*(1.f-gy), gx*(1.f-gy), (1.f-gx)*gy, gx*gy };
    float acc = 0.f;
#pragma unroll
    for (int c = 0; c < 3; ++c) {
        const float* s = img + ((size_t)b * 3 + c) * HW;
        float v = 0.f;
#pragma unroll
        for (int t = 0; t < 4; ++t) {
            int xi = xs[t], yi = ys[t];
            if (xi >= 0 && xi < Ww && yi >= 0 && yi < Hh)
                v += tw[t] * s[(size_t)yi * Ww + xi];
        }
        acc += fabsf(ref[((size_t)b * 3 + c) * HW + hw] - v);
    }
    outM[idx] = acc * (1.f / 3.f);
}

// ---------------------------------------------------------------------------
// Kernel: softsplat scatter (atomicAdd).  wgt = exp(alpha * metric).
// Adds src_c * wgt * tapw into dst channels [coff..coff+C), and wgt * tapw
// into den (once, when writeDen).
// ---------------------------------------------------------------------------
__global__ void splat_scatter_k(const float* __restrict__ src, int C,
                                const float* __restrict__ flow, float flowScale,
                                const float* __restrict__ metric,
                                const float* __restrict__ alphaPtr,
                                float* __restrict__ dst, int Ctot, int coff,
                                float* __restrict__ den, int writeDen,
                                int Hh, int Ww, int Bn)
{
    size_t idx = (size_t)blockIdx.x * THREADS + threadIdx.x;
    int HW = Hh * Ww;
    size_t total = (size_t)Bn * HW;
    if (idx >= total) return;
    int b  = (int)(idx / (size_t)HW);
    int hw = (int)(idx % (size_t)HW);
    int y = hw / Ww, x = hw - y * Ww;
    float a   = alphaPtr[0];
    float wgt = __expf(a * metric[idx]);
    float fx = (float)x + flowScale * flow[((size_t)b * 2 + 0) * HW + hw];
    float fy = (float)y + flowScale * flow[((size_t)b * 2 + 1) * HW + hw];
    float flx = floorf(fx), fly = floorf(fy);
    int x0 = (int)flx, y0 = (int)fly;
    float gx = fx - flx, gy = fy - fly;
    int   xs[4] = { x0, x0 + 1, x0,     x0 + 1 };
    int   ys[4] = { y0, y0,     y0 + 1, y0 + 1 };
    float tw[4] = { (1.f-gx)*(1.f-gy), gx*(1.f-gy), (1.f-gx)*gy, gx*gy };
#pragma unroll
    for (int t = 0; t < 4; ++t) {
        int xi = xs[t], yi = ys[t];
        if (xi < 0 || xi >= Ww || yi < 0 || yi >= Hh) continue;
        float w = wgt * tw[t];
        size_t p = (size_t)yi * Ww + xi;
        for (int c = 0; c < C; ++c) {
            atomicAdd(&dst[((size_t)b * Ctot + coff + c) * HW + p],
                      src[((size_t)b * C + c) * HW + hw] * w);
        }
        if (writeDen) atomicAdd(&den[(size_t)b * HW + p], w);
    }
}

// ---------------------------------------------------------------------------
// Kernel: normalize splat result in place: dst /= (den + eps)
// ---------------------------------------------------------------------------
__global__ void splat_norm_k(float* __restrict__ dst, int Ctot, int coff, int C,
                             const float* __restrict__ den, int HW, int Bn)
{
    size_t idx = (size_t)blockIdx.x * THREADS + threadIdx.x;
    size_t total = (size_t)Bn * C * HW;
    if (idx >= total) return;
    int p = (int)(idx % (size_t)HW);
    size_t t = idx / (size_t)HW;
    int c = (int)(t % (size_t)C);
    int b = (int)(t / (size_t)C);
    float d = den[(size_t)b * HW + p] + 1e-7f;
    size_t o = ((size_t)b * Ctot + coff + c) * HW + p;
    dst[o] = dst[o] / d;
}

// ---------------------------------------------------------------------------
// Host-side orchestration
// ---------------------------------------------------------------------------
static inline unsigned blk(size_t n) { return (unsigned)((n + THREADS - 1) / THREADS); }
static inline int cpad32(int c) { return ((c + 31) / 32) * 32; }

extern "C" void kernel_launch(void* const* d_in, const int* in_sizes, int n_in,
                              void* d_out, int out_size, void* d_ws, size_t ws_size,
                              hipStream_t stream)
{
    (void)in_sizes; (void)n_in; (void)out_size; (void)ws_size;

    const size_t HW  = (size_t)HH * WW_;
    const size_t HWh = (size_t)H2 * W2;
    const size_t HWq = (size_t)H4 * W4;
    const size_t N1  = (size_t)BB * HW;
    const size_t N2  = (size_t)BB * HWh;
    const size_t N4  = (size_t)BB * HWq;

    // ---- inputs (setup_inputs dict order, params dict flattened in order) ----
    const float* img1    = (const float*)d_in[0];
    const float* img2    = (const float*)d_in[1];
    const float* hidden1 = (const float*)d_in[2];
    const float* hidden2 = (const float*)d_in[3];
    const float* flow01  = (const float*)d_in[4];
    const float* flow10  = (const float*)d_in[5];
    const float* alpha   = (const float*)d_in[6];
    const float* fe1_w = (const float*)d_in[7];  const float* fe1_b = (const float*)d_in[8];
    const float* fe2_w = (const float*)d_in[9];  const float* fe2_b = (const float*)d_in[10];
    const float* fe3_w = (const float*)d_in[11]; const float* fe3_b = (const float*)d_in[12];
    const float* m1_w  = (const float*)d_in[13]; const float* m1_b  = (const float*)d_in[14];
    const float* m2_w  = (const float*)d_in[15]; const float* m2_b  = (const float*)d_in[16];
    const float* mix_w = (const float*)d_in[17]; const float* mix_b = (const float*)d_in[18];
    const float* g3_w  = (const float*)d_in[19]; const float* g3_b  = (const float*)d_in[20];
    const float* g2_w  = (const float*)d_in[21]; const float* g2_b  = (const float*)d_in[22];
    const float* g1_w  = (const float*)d_in[23]; const float* g1_b  = (const float*)d_in[24];

    float* out_mix = (float*)d_out;                         // [B,64,H,W]
    float* out_im  = (float*)d_out + (size_t)BB * 64 * HW;  // [B,3,H,W]

    // ---- deterministic workspace layout ----
    uint8_t* base = (uint8_t*)d_ws;
    size_t off = 0;
    auto alloc = [&](size_t bytes) -> void* {
        void* p = base + off;
        off = (off + bytes + 255) & ~(size_t)255;
        return p;
    };
    float* f12_0 = (float*)alloc(2 * N1 * 4);
    float* f12_1 = (float*)alloc(2 * N2 * 4);
    float* f12_2 = (float*)alloc(2 * N4 * 4);
    float* f21_0 = (float*)alloc(2 * N1 * 4);
    float* f21_1 = (float*)alloc(2 * N2 * 4);
    float* f21_2 = (float*)alloc(2 * N4 * 4);
    float* f1a = (float*)alloc(32 * N1 * 4);
    float* f2a = (float*)alloc(64 * N2 * 4);
    float* f3a = (float*)alloc(96 * N4 * 4);
    float* f1b = (float*)alloc(32 * N1 * 4);
    float* f2b = (float*)alloc(64 * N2 * 4);
    float* f3b = (float*)alloc(96 * N4 * 4);
    float* mtmp = (float*)alloc(N1 * 4);
    float* mhid = (float*)alloc(16 * N1 * 4);
    float* m12  = (float*)alloc(N1 * 4);
    float* m12h = (float*)alloc(N2 * 4);
    float* m12q = (float*)alloc(N4 * 4);
    float* m21  = (float*)alloc(N1 * 4);
    float* m21h = (float*)alloc(N2 * 4);
    float* m21q = (float*)alloc(N4 * 4);
    float* den  = (float*)alloc(N1 * 4);
    float* l1 = (float*)alloc(198 * N1 * 4);
    float* l2 = (float*)alloc(128 * N2 * 4);
    float* l3 = (float*)alloc(192 * N4 * 4);
    float* g3o  = (float*)alloc(64 * N4 * 4);
    float* g3up = (float*)alloc(64 * N2 * 4);
    float* g2o  = (float*)alloc(32 * N2 * 4);
    float* g2up = (float*)alloc(32 * N1 * 4);
    _Float16* nhwc = (_Float16*)alloc(256 * N1 * 2);

    auto packAlloc = [&](int Cout, int Cin) -> _Float16* {
        int mt = (Cout + 15) / 16;
        int KC = 9 * (cpad32(Cin) / 32);
        return (_Float16*)alloc((size_t)mt * KC * 512 * 2);
    };
    _Float16* wp_fe1 = packAlloc(32, 3);
    _Float16* wp_fe2 = packAlloc(64, 32);
    _Float16* wp_fe3 = packAlloc(96, 64);
    _Float16* wp_m1  = packAlloc(16, 4);
    _Float16* wp_m2  = packAlloc(1, 16);
    _Float16* wp_mix = packAlloc(64, 128);
    _Float16* wp_g3  = packAlloc(64, 192);
    _Float16* wp_g2  = packAlloc(32, 192);
    _Float16* wp_g1  = packAlloc(3, 230);

    // ---- launch helpers ----
    auto packW = [&](const float* w, _Float16* dst, int Cout, int Cin) {
        int Cinpad = cpad32(Cin);
        int mt = (Cout + 15) / 16;
        int KC = 9 * (Cinpad / 32);
        size_t total = (size_t)mt * KC * 512;
        pack_weights_k<<<blk(total), THREADS, 0, stream>>>(w, dst, Cout, Cin, Cinpad, KC, total);
    };
    auto toNHWC = [&](const float* src, int srcCtot, int srcCoff, int Ccopy,
                      _Float16* dst, int Cpad, int dstCoff, size_t HWn) {
        size_t total = (size_t)BB * HWn * Ccopy;
        nchw_to_nhwc_f16_k<<<blk(total), THREADS, 0, stream>>>(
            src, srcCtot, srcCoff, Ccopy, dst, Cpad, dstCoff, BB, (int)HWn);
    };
    auto conv = [&](const _Float16* in, const _Float16* wpk, const float* bias, float* out,
                    int Hin, int Win, int Hout, int Wout, int Cinpad, int Cout,
                    int stride, int relu) {
        int mt = (Cout + 15) / 16;
        int tiles = (Hout * Wout) / 16;
        dim3 grid((tiles + 7) / 8, mt, BB);
        conv3x3_wmma_k<<<grid, 256, 0, stream>>>(in, wpk, bias, out,
                                                 Hin, Win, Hout, Wout, Cinpad, Cout, stride, relu);
    };
    auto resize = [&](const float* src, float* dst, int C, int Hs, int Ws, int Hd, int Wd, float mul) {
        size_t total = (size_t)BB * C * Hd * Wd;
        resize_bilinear_k<<<blk(total), THREADS, 0, stream>>>(src, dst, C, Hs, Ws, Hd, Wd, BB, mul);
    };
    auto scatter = [&](const float* src, int C, const float* flow, const float* metric,
                       float* dst, int Ctot, int coff, int writeDen, int Hh, int Ww) {
        size_t total = (size_t)BB * Hh * Ww;
        splat_scatter_k<<<blk(total), THREADS, 0, stream>>>(
            src, C, flow, 0.5f, metric, alpha, dst, Ctot, coff, den, writeDen, Hh, Ww, BB);
    };
    auto norm = [&](float* dst, int Ctot, int coff, int C, size_t HWn) {
        size_t total = (size_t)BB * C * HWn;
        splat_norm_k<<<blk(total), THREADS, 0, stream>>>(dst, Ctot, coff, C, den, (int)HWn, BB);
    };
    auto zero = [&](void* p, size_t bytes) { hipMemsetAsync(p, 0, bytes, stream); };

    // ---- 1) pack weights ----
    packW(fe1_w, wp_fe1, 32, 3);
    packW(fe2_w, wp_fe2, 64, 32);
    packW(fe3_w, wp_fe3, 96, 64);
    packW(m1_w,  wp_m1,  16, 4);
    packW(m2_w,  wp_m2,  1, 16);
    packW(mix_w, wp_mix, 64, 128);
    packW(g3_w,  wp_g3,  64, 192);
    packW(g2_w,  wp_g2,  32, 192);
    packW(g1_w,  wp_g1,  3, 230);

    // ---- 2) flow pyramids: 20*up(full), 10*up(half), 5*identity(quarter) ----
    resize(flow01, f12_0, 2, H4, W4, HH, WW_, 20.f);
    resize(flow01, f12_1, 2, H4, W4, H2, W2, 10.f);
    resize(flow01, f12_2, 2, H4, W4, H4, W4, 5.f);
    resize(flow10, f21_0, 2, H4, W4, HH, WW_, 20.f);
    resize(flow10, f21_1, 2, H4, W4, H2, W2, 10.f);
    resize(flow10, f21_2, 2, H4, W4, H4, W4, 5.f);

    // ---- 3) feature pyramids ----
    // img1
    zero(nhwc, N1 * 32 * 2);
    toNHWC(img1, 3, 0, 3, nhwc, 32, 0, HW);
    conv(nhwc, wp_fe1, fe1_b, f1a, HH, WW_, HH, WW_, 32, 32, 1, 1);
    toNHWC(f1a, 32, 0, 32, nhwc, 32, 0, HW);
    conv(nhwc, wp_fe2, fe2_b, f2a, HH, WW_, H2, W2, 32, 64, 2, 1);
    toNHWC(f2a, 64, 0, 64, nhwc, 64, 0, HWh);
    conv(nhwc, wp_fe3, fe3_b, f3a, H2, W2, H4, W4, 64, 96, 2, 1);
    // img2
    zero(nhwc, N1 * 32 * 2);
    toNHWC(img2, 3, 0, 3, nhwc, 32, 0, HW);
    conv(nhwc, wp_fe1, fe1_b, f1b, HH, WW_, HH, WW_, 32, 32, 1, 1);
    toNHWC(f1b, 32, 0, 32, nhwc, 32, 0, HW);
    conv(nhwc, wp_fe2, fe2_b, f2b, HH, WW_, H2, W2, 32, 64, 2, 1);
    toNHWC(f2b, 64, 0, 64, nhwc, 64, 0, HWh);
    conv(nhwc, wp_fe3, fe3_b, f3b, H2, W2, H4, W4, 64, 96, 2, 1);

    // ---- 4) metrics ----
    // m12 = metric_net(mean|img1 - warp(img2, f12_0)|, img1)
    {
        size_t total = N1;
        warp_absdiff_k<<<blk(total), THREADS, 0, stream>>>(img1, img2, f12_0, mtmp, HH, WW_, BB);
    }
    zero(nhwc, N1 * 32 * 2);
    toNHWC(mtmp, 1, 0, 1, nhwc, 32, 0, HW);
    toNHWC(img1, 3, 0, 3, nhwc, 32, 1, HW);
    conv(nhwc, wp_m1, m1_b, mhid, HH, WW_, HH, WW_, 32, 16, 1, 1);
    zero(nhwc, N1 * 32 * 2);
    toNHWC(mhid, 16, 0, 16, nhwc, 32, 0, HW);
    conv(nhwc, wp_m2, m2_b, m12, HH, WW_, HH, WW_, 32, 1, 1, 0);
    resize(m12, m12h, 1, HH, WW_, H2, W2, 1.f);
    resize(m12, m12q, 1, HH, WW_, H4, W4, 1.f);
    // m21
    {
        size_t total = N1;
        warp_absdiff_k<<<blk(total), THREADS, 0, stream>>>(img2, img1, f21_0, mtmp, HH, WW_, BB);
    }
    zero(nhwc, N1 * 32 * 2);
    toNHWC(mtmp, 1, 0, 1, nhwc, 32, 0, HW);
    toNHWC(img2, 3, 0, 3, nhwc, 32, 1, HW);
    conv(nhwc, wp_m1, m1_b, mhid, HH, WW_, HH, WW_, 32, 16, 1, 1);
    zero(nhwc, N1 * 32 * 2);
    toNHWC(mhid, 16, 0, 16, nhwc, 32, 0, HW);
    conv(nhwc, wp_m2, m2_b, m21, HH, WW_, HH, WW_, 32, 1, 1, 0);
    resize(m21, m21h, 1, HH, WW_, H2, W2, 1.f);
    resize(m21, m21q, 1, HH, WW_, H4, W4, 1.f);

    // ---- 5) softsplat into concat buffers ----
    // full res: l1 = [wi1(3), wp1_1(32), wh1(64), wi2(3), wp2_1(32), wh2(64)]
    zero(l1, 198 * N1 * 4);
    zero(den, N1 * 4);
    scatter(img1,    3,  f12_0, m12, l1, 198, 0,   1, HH, WW_);
    scatter(f1a,     32, f12_0, m12, l1, 198, 3,   0, HH, WW_);
    scatter(hidden1, 64, f12_0, m12, l1, 198, 35,  0, HH, WW_);
    norm(l1, 198, 0, 99, HW);
    zero(den, N1 * 4);
    scatter(img2,    3,  f21_0, m21, l1, 198, 99,  1, HH, WW_);
    scatter(f1b,     32, f21_0, m21, l1, 198, 102, 0, HH, WW_);
    scatter(hidden2, 64, f21_0, m21, l1, 198, 134, 0, HH, WW_);
    norm(l1, 198, 99, 99, HW);
    // half res: l2 = [wp1_2(64), wp2_2(64)]
    zero(l2, 128 * N2 * 4);
    zero(den, N2 * 4);
    scatter(f2a, 64, f12_1, m12h, l2, 128, 0, 1, H2, W2);
    norm(l2, 128, 0, 64, HWh);
    zero(den, N2 * 4);
    scatter(f2b, 64, f21_1, m21h, l2, 128, 64, 1, H2, W2);
    norm(l2, 128, 64, 64, HWh);
    // quarter res: l3 = [wp1_3(96), wp2_3(96)]
    zero(l3, 192 * N4 * 4);
    zero(den, N4 * 4);
    scatter(f3a, 96, f12_2, m12q, l3, 192, 0, 1, H4, W4);
    norm(l3, 192, 0, 96, HWq);
    zero(den, N4 * 4);
    scatter(f3b, 96, f21_2, m21q, l3, 192, 96, 1, H4, W4);
    norm(l3, 192, 96, 96, HWq);

    // ---- 6) out_feat_mix = conv(concat[wh1, wh2]) ----
    toNHWC(l1, 198, 35, 64,  nhwc, 128, 0,  HW);
    toNHWC(l1, 198, 134, 64, nhwc, 128, 64, HW);
    conv(nhwc, wp_mix, mix_b, out_mix, HH, WW_, HH, WW_, 128, 64, 1, 0);

    // ---- 7) GridNet decoder ----
    toNHWC(l3, 192, 0, 192, nhwc, 192, 0, HWq);
    conv(nhwc, wp_g3, g3_b, g3o, H4, W4, H4, W4, 192, 64, 1, 1);
    resize(g3o, g3up, 64, H4, W4, H2, W2, 1.f);

    toNHWC(l2, 128, 0, 128,  nhwc, 192, 0,   HWh);
    toNHWC(g3up, 64, 0, 64,  nhwc, 192, 128, HWh);
    conv(nhwc, wp_g2, g2_b, g2o, H2, W2, H2, W2, 192, 32, 1, 1);
    resize(g2o, g2up, 32, H2, W2, HH, WW_, 1.f);

    zero(nhwc, N1 * 256 * 2);
    toNHWC(l1, 198, 0, 198,  nhwc, 256, 0,   HW);
    toNHWC(g2up, 32, 0, 32,  nhwc, 256, 198, HW);
    conv(nhwc, wp_g1, g1_b, out_im, HH, WW_, HH, WW_, 256, 3, 1, 0);
}
